// HetEncoder_58858231824572
// MI455X (gfx1250) — compile-verified, matrix operand order checked by reading, not run on hardware
//
#include <hip/hip_runtime.h>

typedef __attribute__((ext_vector_type(16))) _Float16 v16h;
typedef __attribute__((ext_vector_type(8)))  float    v8f;

#define THREADS 256

enum { SEL_A = 0, SEL_I = 1, SEL_EA = 2, SEL_EI = 3, SEL_ONE = 4 };

union HFrag { v16h v; uint4 q[2]; };

__device__ __forceinline__ int sel_widx(int c, int selKind) {
  int ta = c / 3, ti = c % 3;
  switch (selKind) {
    case SEL_A:  return ta;
    case SEL_I:  return ti;
    case SEL_EA: return ti * 3 + ta;   // item -> anchor edge
    case SEL_EI: return c;             // anchor -> item edge
    default:     return 0;
  }
}

// ---------------------------------------------------------------- bucketing
__global__ __launch_bounds__(THREADS) void init_meta_kernel(int* perm, int P, int* meta) {
  int idx = blockIdx.x * THREADS + threadIdx.x;
  if (idx < P) perm[idx] = -1;
  if (idx < 64) meta[idx] = 0;
}

__global__ __launch_bounds__(THREADS) void hist_kernel(const int* __restrict__ at,
                                                       const int* __restrict__ it,
                                                       int* __restrict__ cnt, int B) {
  int idx = blockIdx.x * THREADS + threadIdx.x;
  if (idx >= B) return;
  atomicAdd(&cnt[at[idx] * 3 + it[idx]], 1);
}

__global__ void scan_kernel(int* meta) {
  // cnt = meta[0..8], cursors = meta[9..17], off = meta[18..27]
  if (threadIdx.x == 0) {
    int acc = 0;
    meta[18] = 0;
    for (int c = 0; c < 9; ++c) {
      acc += (meta[c] + 63) & ~63;      // pad each bucket to 64-row tiles
      meta[18 + c + 1] = acc;
    }
  }
}

__global__ __launch_bounds__(THREADS) void scatter_kernel(const int* __restrict__ at,
                                                          const int* __restrict__ it,
                                                          int* meta, int* perm, int B) {
  int idx = blockIdx.x * THREADS + threadIdx.x;
  if (idx >= B) return;
  int c = at[idx] * 3 + it[idx];
  int pos = meta[18 + c] + atomicAdd(&meta[9 + c], 1);
  perm[pos] = idx;
}

// ------------------------------------------- weight transpose + f32->f16
// src: [nmat][Kin][256] f32 ; dst: [nmat][256][Kpad] f16 (n-major, zero-padded K)
__global__ __launch_bounds__(THREADS) void wconv_kernel(const float* __restrict__ src,
                                                        _Float16* __restrict__ dst,
                                                        int nmat, int Kin, int Kpad) {
  long long idx = (long long)blockIdx.x * THREADS + threadIdx.x;
  long long total = (long long)nmat * 256 * Kpad;
  if (idx >= total) return;
  int kcol = (int)(idx % Kpad);
  int n = (int)((idx / Kpad) % 256);
  int m = (int)(idx / ((long long)Kpad * 256));
  float x = (kcol < Kin) ? src[((long long)m * Kin + kcol) * 256 + n] : 0.f;
  dst[idx] = (_Float16)x;
}

// -------------------------------------- gather features (padded) -> f16
__global__ __launch_bounds__(THREADS) void gather_feats_kernel(const float* __restrict__ af,
                                                               const float* __restrict__ itf,
                                                               const int* __restrict__ perm,
                                                               const int* __restrict__ offp,
                                                               _Float16* __restrict__ Xa,
                                                               _Float16* __restrict__ Xi) {
  int idx = blockIdx.x * THREADS + threadIdx.x;
  int row = idx >> 6, d = idx & 63;
  if (row >= offp[9]) return;
  int src = perm[row];
  float va = 0.f, vi = 0.f;
  if (src >= 0 && d < 60) { va = af[src * 60 + d]; vi = itf[src * 60 + d]; }
  Xa[idx] = (_Float16)va;
  Xi[idx] = (_Float16)vi;
}

// ---------------------------------------------------------- async -> LDS
__device__ __forceinline__ void async_copy_to_lds(const _Float16* src, _Float16* lds,
                                                  int halves, int tid) {
  unsigned ldsBase = (unsigned)(uintptr_t)lds;
  int nchunks = halves >> 3;                  // 16-byte chunks
  for (int ci = tid; ci < nchunks; ci += THREADS) {
    unsigned laddr = ldsBase + ci * 16;
    unsigned long long ga = (unsigned long long)(uintptr_t)(src + ci * 8);
    asm volatile("global_load_async_to_lds_b128 %0, %1, off"
                 :: "v"(laddr), "v"(ga) : "memory");
  }
}

// -------------------------------------------- LDS-staged WMMA GEMM (k/q/v)
// OutH[64x128 tile] = A16[64x256] @ W[sel][256x(n0..n0+127)], f16 out, K=256.
// Both operands async-staged into LDS; inner loop = ds_load + wmma only.
// grid = 2 * (P/64); blockIdx: row tile = bid>>1, col half = bid&1.
__global__ __launch_bounds__(THREADS) void hgemm_lds_kernel(
    const _Float16* __restrict__ A16, const _Float16* __restrict__ Wt,
    _Float16* __restrict__ OutH, const int* __restrict__ offp, int selKind) {
  const int K = 256;
  __shared__ __align__(16) _Float16 As[64 * 256];    // 32 KB
  __shared__ __align__(16) _Float16 Ws[128 * 256];   // 64 KB

  int m0 = (blockIdx.x >> 1) * 64;
  int n0 = (blockIdx.x & 1) * 128;
  if (m0 >= offp[9]) return;                 // block-uniform
  int c = 0;
  while (c < 8 && m0 >= offp[c + 1]) ++c;
  int widx = sel_widx(c, selKind);
  const _Float16* W = Wt + ((size_t)widx * 256 + n0) * K;  // 128 rows, contiguous

  int tid = threadIdx.x;
  async_copy_to_lds(A16 + (size_t)m0 * K, As, 64 * K, tid);
  async_copy_to_lds(W, Ws, 128 * K, tid);
  asm volatile("s_wait_asynccnt 0x0" ::: "memory");
  __syncthreads();

  int wave = tid >> 5, lane = tid & 31;
  int laneLo = lane & 15;
  int hiHalf = lane >> 4;
  int mbase = (wave >> 1) * 16;     // 4 row groups
  int nloc = (wave & 1) * 64;       // 2 col halves of 64
  int kSel = hiHalf * 8;

  v8f acc[4];
#pragma unroll
  for (int nt = 0; nt < 4; ++nt)
#pragma unroll
    for (int r = 0; r < 8; ++r) acc[nt][r] = 0.f;

  for (int kk = 0; kk < K; kk += 32) {
    HFrag af;
    const _Float16* ap = &As[(mbase + laneLo) * K + kk + kSel];
    af.q[0] = *(const uint4*)(ap);
    af.q[1] = *(const uint4*)(ap + 16);
    HFrag bf[4];
#pragma unroll
    for (int nt = 0; nt < 4; ++nt) {
      const _Float16* bp = &Ws[(nloc + nt * 16 + laneLo) * K + kk + kSel];
      bf[nt].q[0] = *(const uint4*)(bp);
      bf[nt].q[1] = *(const uint4*)(bp + 16);
    }
#pragma unroll
    for (int nt = 0; nt < 4; ++nt)
      acc[nt] = __builtin_amdgcn_wmma_f32_16x16x32_f16(
          false, af.v, false, bf[nt].v, (short)0, acc[nt], false, false);
  }

#pragma unroll
  for (int nt = 0; nt < 4; ++nt) {
    int col = n0 + nloc + nt * 16 + laneLo;
#pragma unroll
    for (int r = 0; r < 8; ++r) {
      int row = m0 + mbase + r + hiHalf * 8;
      OutH[(size_t)row * 256 + col] = (_Float16)acc[nt][r];
    }
  }
}

// ------------------------------------------------------------- WMMA GEMM
// (optional LN)(A16[64xK] @ W[sel][Kx256] + bias + resid), relu
// writes OutF (f32, nullable) and OutH (f16, nullable)
// block = 256 threads = 8 waves; wave w: rows 16*(w/2).., cols 128*(w&1)..
__global__ __launch_bounds__(THREADS) void hgemm_kernel(
    const _Float16* __restrict__ A16, const _Float16* __restrict__ Wt,
    const float* __restrict__ bias, const float* __restrict__ resid,
    const float* __restrict__ gamma, const float* __restrict__ beta,
    float* __restrict__ OutF, _Float16* __restrict__ OutH,
    const int* __restrict__ offp, int K, int kshift, int selKind, int relu) {
  __shared__ __align__(16) _Float16 As[64 * 256];
  __shared__ float redS[64][2];
  __shared__ float redQ[64][2];

  int m0 = blockIdx.x * 64;
  if (m0 >= offp[9]) return;                 // block-uniform -> EXEC stays full
  int c = 0;
  while (c < 8 && m0 >= offp[c + 1]) ++c;    // bucket of this 64-row tile
  int widx = sel_widx(c, selKind);
  const _Float16* W = Wt + (size_t)widx * 256 * K;

  int tid = threadIdx.x;
  async_copy_to_lds(A16 + ((size_t)m0 << kshift), As, 64 << kshift, tid);
  asm volatile("s_wait_asynccnt 0x0" ::: "memory");
  __syncthreads();

  int wave = tid >> 5, lane = tid & 31;
  int laneLo = lane & 15;
  int hiHalf = lane >> 4;              // 0: K{0..7,16..23}  1: K{8..15,24..31}
  int mbase = (wave >> 1) * 16;
  int nbase = (wave & 1) * 128;
  int kSel = hiHalf * 8;

  v8f acc[8];
#pragma unroll
  for (int nt = 0; nt < 8; ++nt)
#pragma unroll
    for (int r = 0; r < 8; ++r) acc[nt][r] = 0.f;

  for (int kk = 0; kk < K; kk += 32) {
    HFrag af;
    const _Float16* ap = &As[((mbase + laneLo) << kshift) + kk + kSel];
    af.q[0] = *(const uint4*)(ap);
    af.q[1] = *(const uint4*)(ap + 16);
    if (kk + 32 < K)
      __builtin_prefetch(W + (size_t)(nbase + laneLo) * K + kk + 32, 0, 1);
    const _Float16* bbase = W + (size_t)kk + kSel;
    HFrag bf[4];
#pragma unroll
    for (int p = 0; p < 3; ++p) {
      const _Float16* bp = bbase + (size_t)(nbase + p * 16 + laneLo) * K;
      bf[p].q[0] = *(const uint4*)(bp);
      bf[p].q[1] = *(const uint4*)(bp + 16);
    }
#pragma unroll
    for (int nt = 0; nt < 8; ++nt) {
      if (nt < 5) {
        const _Float16* bp = bbase + (size_t)(nbase + (nt + 3) * 16 + laneLo) * K;
        bf[(nt + 3) & 3].q[0] = *(const uint4*)(bp);
        bf[(nt + 3) & 3].q[1] = *(const uint4*)(bp + 16);
      }
      acc[nt] = __builtin_amdgcn_wmma_f32_16x16x32_f16(
          false, af.v, false, bf[nt & 3].v, (short)0, acc[nt], false, false);
    }
  }

  // epilogue: bias + residual
#pragma unroll
  for (int nt = 0; nt < 8; ++nt) {
    int col = nbase + nt * 16 + laneLo;
    float bcol = bias ? bias[widx * 256 + col] : 0.f;
#pragma unroll
    for (int r = 0; r < 8; ++r) {
      int row = m0 + mbase + r + hiHalf * 8;
      float x = acc[nt][r] + bcol;
      if (resid) x += resid[(size_t)row * 256 + col];
      acc[nt][r] = x;
    }
  }

  if (gamma) {  // LayerNorm across the 256 cols of each row
    float ps[8], pq[8];
#pragma unroll
    for (int r = 0; r < 8; ++r) { ps[r] = 0.f; pq[r] = 0.f; }
#pragma unroll
    for (int nt = 0; nt < 8; ++nt)
#pragma unroll
      for (int r = 0; r < 8; ++r) { float x = acc[nt][r]; ps[r] += x; pq[r] += x * x; }
#pragma unroll
    for (int m = 1; m < 16; m <<= 1) {
#pragma unroll
      for (int r = 0; r < 8; ++r) {
        ps[r] += __shfl_xor(ps[r], m, 32);
        pq[r] += __shfl_xor(pq[r], m, 32);
      }
    }
    if (laneLo == 0) {
#pragma unroll
      for (int r = 0; r < 8; ++r) {
        int row = mbase + r + hiHalf * 8;
        redS[row][wave & 1] = ps[r];
        redQ[row][wave & 1] = pq[r];
      }
    }
    __syncthreads();
    float mu[8], rs[8];
#pragma unroll
    for (int r = 0; r < 8; ++r) {
      int row = mbase + r + hiHalf * 8;
      float s = redS[row][0] + redS[row][1];
      float q2 = redQ[row][0] + redQ[row][1];
      float m = s * (1.f / 256.f);
      float var = q2 * (1.f / 256.f) - m * m;
      mu[r] = m;
      rs[r] = rsqrtf(var + 1e-5f);
    }
#pragma unroll
    for (int nt = 0; nt < 8; ++nt) {
      int col = nbase + nt * 16 + laneLo;
      float g = gamma[widx * 256 + col];
      float be = beta[widx * 256 + col];
#pragma unroll
      for (int r = 0; r < 8; ++r)
        acc[nt][r] = (acc[nt][r] - mu[r]) * rs[r] * g + be;
    }
  }

#pragma unroll
  for (int nt = 0; nt < 8; ++nt) {
    int col = nbase + nt * 16 + laneLo;
#pragma unroll
    for (int r = 0; r < 8; ++r) {
      int row = m0 + mbase + r + hiHalf * 8;
      float y = acc[nt][r];
      if (relu) y = fmaxf(y, 0.f);
      if (OutF) OutF[(size_t)row * 256 + col] = y;
      if (OutH) OutH[(size_t)row * 256 + col] = (_Float16)y;
    }
  }
}

// ------------------------------- per-head sigmoid gate (wave per row, f16)
__global__ __launch_bounds__(THREADS) void attn_kernel(const _Float16* __restrict__ q16,
                                                       const _Float16* __restrict__ k16,
                                                       const _Float16* __restrict__ v16,
                                                       _Float16* __restrict__ msg16,
                                                       const int* __restrict__ offp) {
  int row = blockIdx.x * 8 + (threadIdx.x >> 5);
  if (row >= offp[9]) return;
  int lane = threadIdx.x & 31;
  size_t base = (size_t)row * 256 + lane;   // lane-strided: fully coalesced
  float qv[8], kv[8], vv[8];
#pragma unroll
  for (int j = 0; j < 8; ++j) {
    qv[j] = (float)q16[base + j * 32];
    kv[j] = (float)k16[base + j * 32];
    vv[j] = (float)v16[base + j * 32];
  }
  // cols j*32+lane -> head = j>>1 (4 heads x 64 dims)
  float d[4];
#pragma unroll
  for (int h = 0; h < 4; ++h)
    d[h] = qv[2 * h] * kv[2 * h] + qv[2 * h + 1] * kv[2 * h + 1];
#pragma unroll
  for (int m = 1; m < 32; m <<= 1)
#pragma unroll
    for (int h = 0; h < 4; ++h) d[h] += __shfl_xor(d[h], m, 32);
  float at[4];
#pragma unroll
  for (int h = 0; h < 4; ++h)
    at[h] = 1.f / (1.f + expf(-d[h] * 0.125f));   // SCALE = sqrt(64) = 8
#pragma unroll
  for (int j = 0; j < 8; ++j)
    msg16[base + j * 32] = (_Float16)(at[j >> 1] * vv[j]);
}

// -------------------------------------------- L2 normalize + scatter back
__global__ __launch_bounds__(THREADS) void l2_scatter_kernel(const float* __restrict__ X,
                                                             const int* __restrict__ perm,
                                                             const int* __restrict__ offp,
                                                             float* __restrict__ out) {
  int row = blockIdx.x * 8 + (threadIdx.x >> 5);  // one wave per row
  if (row >= offp[9]) return;
  int lane = threadIdx.x & 31;
  int orow = perm[row];
  if (orow < 0) return;                            // padding row
  float vals[8];
  float s2 = 0.f;
#pragma unroll
  for (int j = 0; j < 8; ++j) {
    float x = X[(size_t)row * 256 + j * 32 + lane];
    vals[j] = x;
    s2 += x * x;
  }
#pragma unroll
  for (int m = 1; m < 32; m <<= 1) s2 += __shfl_xor(s2, m, 32);
  float rn = 1.f / fmaxf(sqrtf(s2), 1e-12f);
#pragma unroll
  for (int j = 0; j < 8; ++j)
    out[(size_t)orow * 256 + j * 32 + lane] = vals[j] * rn;
}

// ---------------------------------------------------------------- driver
extern "C" void kernel_launch(void* const* d_in, const int* in_sizes, int n_in,
                              void* d_out, int out_size, void* d_ws, size_t ws_size,
                              hipStream_t stream) {
  const float* anchor_feats = (const float*)d_in[0];
  const float* item_feats   = (const float*)d_in[1];
  const int*   atid         = (const int*)d_in[2];
  const int*   itid         = (const int*)d_in[3];
  const float* enc_W1  = (const float*)d_in[4];
  const float* enc_b1  = (const float*)d_in[5];
  const float* enc_g1  = (const float*)d_in[6];
  const float* enc_be1 = (const float*)d_in[7];
  const float* enc_W2  = (const float*)d_in[8];
  const float* enc_b2  = (const float*)d_in[9];
  const float* enc_g2  = (const float*)d_in[10];
  const float* enc_be2 = (const float*)d_in[11];
  const float* hgt_Wk   = (const float*)d_in[12];
  const float* hgt_Wq   = (const float*)d_in[13];
  const float* hgt_Wv   = (const float*)d_in[14];
  const float* hgt_Wout = (const float*)d_in[15];
  const float* hgt_g    = (const float*)d_in[16];
  const float* hgt_b    = (const float*)d_in[17];
  const float* out_W  = (const float*)d_in[18];
  const float* out_b  = (const float*)d_in[19];
  const float* out_g  = (const float*)d_in[20];
  const float* out_be = (const float*)d_in[21];
  (void)n_in; (void)out_size; (void)ws_size;

  const int B = in_sizes[0] / 60;
  const int P = ((B + 9 * 64) + 63) & ~63;   // padded row capacity

  char* w = (char*)d_ws;
  auto alloc = [&](size_t bytes) -> void* {
    void* p = (void*)w;
    w += (bytes + 255) & ~(size_t)255;
    return p;
  };
  int* perm = (int*)alloc((size_t)P * 4);
  int* meta = (int*)alloc(64 * 4);
  int* offp = meta + 18;
  _Float16* W1t = (_Float16*)alloc((size_t)3 * 256 * 64 * 2);
  _Float16* W2t = (_Float16*)alloc((size_t)3 * 256 * 256 * 2);
  _Float16* Wkt = (_Float16*)alloc((size_t)18 * 256 * 256 * 2);
  _Float16* Wqt = (_Float16*)alloc((size_t)18 * 256 * 256 * 2);
  _Float16* Wvt = (_Float16*)alloc((size_t)18 * 256 * 256 * 2);
  _Float16* Wot = (_Float16*)alloc((size_t)6 * 256 * 256 * 2);
  _Float16* oWt = (_Float16*)alloc((size_t)256 * 256 * 2);
  // f16 activations (GEMM A operands / attention operands)
  _Float16* Xa16  = (_Float16*)alloc((size_t)P * 64 * 2);
  _Float16* Xi16  = (_Float16*)alloc((size_t)P * 64 * 2);
  _Float16* z16   = (_Float16*)alloc((size_t)P * 256 * 2);
  _Float16* ha16  = (_Float16*)alloc((size_t)P * 256 * 2);
  _Float16* hi16  = (_Float16*)alloc((size_t)P * 256 * 2);
  _Float16* haN16 = (_Float16*)alloc((size_t)P * 256 * 2);
  _Float16* hiN16 = (_Float16*)alloc((size_t)P * 256 * 2);
  _Float16* mb16  = (_Float16*)alloc((size_t)P * 256 * 2);
  _Float16* kb16  = (_Float16*)alloc((size_t)P * 256 * 2);
  _Float16* qb16  = (_Float16*)alloc((size_t)P * 256 * 2);
  _Float16* vb16  = (_Float16*)alloc((size_t)P * 256 * 2);
  // f32 activations (residual / final head)
  float* ha  = (float*)alloc((size_t)P * 256 * 4);
  float* hi  = (float*)alloc((size_t)P * 256 * 4);
  float* haN = (float*)alloc((size_t)P * 256 * 4);
  float* hiN = (float*)alloc((size_t)P * 256 * 4);

  dim3 blk(THREADS);
  int gB = (B + THREADS - 1) / THREADS;
  int gP = (P + THREADS - 1) / THREADS;

  init_meta_kernel<<<gP, blk, 0, stream>>>(perm, P, meta);
  hist_kernel<<<gB, blk, 0, stream>>>(atid, itid, meta, B);
  scan_kernel<<<1, 32, 0, stream>>>(meta);
  scatter_kernel<<<gB, blk, 0, stream>>>(atid, itid, meta, perm, B);

  auto wcv = [&](const float* src, _Float16* dst, int nmat, int Kin, int Kpad) {
    long long total = (long long)nmat * 256 * Kpad;
    wconv_kernel<<<(int)((total + THREADS - 1) / THREADS), blk, 0, stream>>>(src, dst, nmat, Kin, Kpad);
  };
  wcv(enc_W1, W1t, 3, 60, 64);
  wcv(enc_W2, W2t, 3, 256, 256);
  wcv(hgt_Wk, Wkt, 18, 256, 256);
  wcv(hgt_Wq, Wqt, 18, 256, 256);
  wcv(hgt_Wv, Wvt, 18, 256, 256);
  wcv(hgt_Wout, Wot, 6, 256, 256);
  wcv(out_W, oWt, 1, 256, 256);

  gather_feats_kernel<<<(P * 64 + THREADS - 1) / THREADS, blk, 0, stream>>>(
      anchor_feats, item_feats, perm, offp, Xa16, Xi16);

  int gG = P / 64;          // N256 LN-GEMM grid
  int gG2 = 2 * gG;         // N128 LDS-GEMM grid
  // encoder: anchors
  hgemm_kernel<<<gG, blk, 0, stream>>>(Xa16, W1t, enc_b1, nullptr, enc_g1, enc_be1,
                                       nullptr, z16, offp, 64, 6, SEL_A, 1);
  hgemm_kernel<<<gG, blk, 0, stream>>>(z16, W2t, enc_b2, nullptr, enc_g2, enc_be2,
                                       ha, ha16, offp, 256, 8, SEL_A, 1);
  // encoder: items
  hgemm_kernel<<<gG, blk, 0, stream>>>(Xi16, W1t, enc_b1, nullptr, enc_g1, enc_be1,
                                       nullptr, z16, offp, 64, 6, SEL_I, 1);
  hgemm_kernel<<<gG, blk, 0, stream>>>(z16, W2t, enc_b2, nullptr, enc_g2, enc_be2,
                                       hi, hi16, offp, 256, 8, SEL_I, 1);

  int gRow = (P + 7) / 8;   // wave-per-row kernels
  float *curA = ha, *curI = hi, *nxtA = haN, *nxtI = hiN;
  _Float16 *curA16 = ha16, *curI16 = hi16, *nxtA16 = haN16, *nxtI16 = hiN16;
  for (int L = 0; L < 2; ++L) {
    size_t wOff = (size_t)L * 9 * 256 * 256;
    size_t oOff = (size_t)L * 3 * 256 * 256;
    const float* gL = hgt_g + L * 3 * 256;
    const float* bL = hgt_b + L * 3 * 256;
    // anchor update: src=item, dst=anchor, e = i*3+a, dt = a
    hgemm_lds_kernel<<<gG2, blk, 0, stream>>>(curI16, Wkt + wOff, kb16, offp, SEL_EA);
    hgemm_lds_kernel<<<gG2, blk, 0, stream>>>(curA16, Wqt + wOff, qb16, offp, SEL_EA);
    hgemm_lds_kernel<<<gG2, blk, 0, stream>>>(curI16, Wvt + wOff, vb16, offp, SEL_EA);
    attn_kernel<<<gRow, blk, 0, stream>>>(qb16, kb16, vb16, mb16, offp);
    hgemm_kernel<<<gG, blk, 0, stream>>>(mb16, Wot + oOff, nullptr, curA, gL, bL,
                                         nxtA, nxtA16, offp, 256, 8, SEL_A, 0);
    // item update: src=anchor, dst=item, e = a*3+i, dt = i
    hgemm_lds_kernel<<<gG2, blk, 0, stream>>>(curA16, Wkt + wOff, kb16, offp, SEL_EI);
    hgemm_lds_kernel<<<gG2, blk, 0, stream>>>(curI16, Wqt + wOff, qb16, offp, SEL_EI);
    hgemm_lds_kernel<<<gG2, blk, 0, stream>>>(curA16, Wvt + wOff, vb16, offp, SEL_EI);
    attn_kernel<<<gRow, blk, 0, stream>>>(qb16, kb16, vb16, mb16, offp);
    hgemm_kernel<<<gG, blk, 0, stream>>>(mb16, Wot + oOff, nullptr, curI, gL, bL,
                                         nxtI, nxtI16, offp, 256, 8, SEL_I, 0);
    float* t;
    t = curA; curA = nxtA; nxtA = t;
    t = curI; curI = nxtI; nxtI = t;
    _Float16* th;
    th = curA16; curA16 = nxtA16; nxtA16 = th;
    th = curI16; curI16 = nxtI16; nxtI16 = th;
  }

  // output head: reuse retired f32 ping-pong buffers as scratch
  float* outF = (float*)d_out;
  hgemm_kernel<<<gG, blk, 0, stream>>>(curA16, oWt, out_b, nullptr, out_g, out_be,
                                       nxtA, nullptr, offp, 256, 8, SEL_ONE, 0);
  l2_scatter_kernel<<<gRow, blk, 0, stream>>>(nxtA, perm, offp, outF);
  hgemm_kernel<<<gG, blk, 0, stream>>>(curI16, oWt, out_b, nullptr, out_g, out_be,
                                       nxtI, nullptr, offp, 256, 8, SEL_ONE, 0);
  l2_scatter_kernel<<<gRow, blk, 0, stream>>>(nxtI, perm, offp, outF + (size_t)B * 256);
}
